// Interval_Refine_75788992905526
// MI455X (gfx1250) — matrix operand order repaired and checked by reading, compiled, not run on hardware
//
#include <hip/hip_runtime.h>
#include <stdint.h>

// ---------------------------------------------------------------------------
// Types for WMMA (gfx1250, wave32)
// ---------------------------------------------------------------------------
typedef __attribute__((ext_vector_type(16))) __bf16 v16bf;
typedef __attribute__((ext_vector_type(8)))  __bf16 v8bf;
typedef __attribute__((ext_vector_type(8)))  float  v8f;

#define DIM   256
#define MAXT  512
#define KPAD  544     // 17 * 32 : padded feature K (519 -> 544)
#define NPAD3 176     // 11 * 16 : padded head output

__device__ __forceinline__ v8f zero8() {
  v8f z = {0.f,0.f,0.f,0.f,0.f,0.f,0.f,0.f};
  return z;
}
__device__ __forceinline__ v16bf join16(v8bf lo, v8bf hi) {
  return __builtin_shufflevector(lo, hi, 0,1,2,3,4,5,6,7,8,9,10,11,12,13,14,15);
}
__device__ __forceinline__ float fsig(float x)  { return 1.0f / (1.0f + __expf(-x)); }
__device__ __forceinline__ float ftanh(float x) { return 2.0f / (1.0f + __expf(-2.0f * x)) - 1.0f; }

// ---------------------------------------------------------------------------
// 1) init: zero feature buffer (bf16) and max-abs slots
// ---------------------------------------------------------------------------
__global__ void k_init(unsigned short* feat, unsigned* mx) {
  int i = blockIdx.x * blockDim.x + threadIdx.x;
  if (i < 80 * KPAD) feat[i] = 0;     // +0.0 bf16
  if (i < 9) mx[i] = 0u;
}

// ---------------------------------------------------------------------------
// 2) prep: per-interval mask, stable compaction, eff count, scalar features
// ---------------------------------------------------------------------------
__global__ void k_prep(const float* tp, const float* npred, const float* alen,
                       const float* anchors, int* idxbuf, int* cnts,
                       unsigned short* feat) {
  __shared__ int scnt[256];
  __shared__ int soff[256];
  int i   = blockIdx.x;        // interval 0..69
  int tid = threadIdx.x;
  float al = alen[0];
  float s = anchors[2*i], e = anchors[2*i+1];

  int c = 0; int loc[4];
  #pragma unroll
  for (int q = 0; q < 4; ++q) {
    int j = tid * 4 + q;
    float t = tp[j] * al;
    bool m = (t >= s) && (t <= e);
    loc[q] = m ? 1 : 0; c += loc[q];
  }
  scnt[tid] = c; __syncthreads();
  if (tid == 0) { int acc = 0; for (int k = 0; k < 256; ++k) { soff[k] = acc; acc += scnt[k]; } scnt[0] = acc; }
  __syncthreads();
  int total = scnt[0];
  int off = soff[tid];
  #pragma unroll
  for (int q = 0; q < 4; ++q) {
    if (loc[q]) { if (off < MAXT) idxbuf[i * MAXT + off] = tid * 4 + q; off++; }
  }
  if (tid == 0) {
    int cntS = total > MAXT ? MAXT : total;
    cnts[i] = cntS;
    int si = i < 15 ? 0 : (i < 55 ? 1 : 2);
    int li = i - (si == 0 ? 0 : (si == 1 ? 15 : 55));
    int rowBase = si == 0 ? 0 : (si == 1 ? 16 : 64);
    float mxv = -1e30f;
    for (int cc = 0; cc < 5; ++cc) { float v = npred[li*5+cc]; mxv = v > mxv ? v : mxv; }
    float sum = 0.f, e0 = 0.f;
    for (int cc = 0; cc < 5; ++cc) { float v = __expf(npred[li*5+cc]-mxv); sum += v; if (cc == 0) e0 = v; }
    __bf16* fr = (__bf16*)(feat + (size_t)(rowBase + li) * KPAD);
    fr[512] = (__bf16)(e0 / sum);
    fr[513] = (__bf16)((s + e) * 0.5f / al);
    fr[514] = (__bf16)((e - s) / al);
  }
}

// ---------------------------------------------------------------------------
// 3) gather: x[i][t][:] = (t < cnt) ? bf16(emb[idx[t]]) : 0
// ---------------------------------------------------------------------------
__global__ void k_gather(const float* emb, const int* idxbuf, const int* cnts,
                         unsigned short* x) {
  int row = blockIdx.x;            // i*512 + t
  int i = row >> 9, t = row & 511;
  int tid = threadIdx.x;
  float v = 0.f;
  if (t < cnts[i]) v = emb[(size_t)idxbuf[row] * DIM + tid];
  ((__bf16*)x)[(size_t)row * DIM + tid] = (__bf16)v;
}

// ---------------------------------------------------------------------------
// 4a) pack GRU weights -> bf16 WMMA B-tile layout.
// B tile (16N x 32K): lane l holds col n=l&15; elems e=0..15 hold
// K = kt*32 + (l>=16 ? 8 : 0) + (e&7) + (e>=8 ? 16 : 0). Per-lane contiguous 32B.
// Layout flat: [mat(0=wih,1=whh)][si*2+dir] x ((nt*8+kt)*32+lane)*16+e
// ---------------------------------------------------------------------------
__global__ void k_packGru(const float* wih, const float* whh, unsigned short* dst) {
  size_t p = (size_t)blockIdx.x * blockDim.x + threadIdx.x;
  if (p >= (size_t)12 * 768 * 256) return;
  int e = p & 15; int lane = (p >> 4) & 31; size_t tix = p >> 9;
  int kt = tix & 7; int nt = (int)((tix >> 3) % 48); int md = (int)(tix / (8 * 48));
  int mat = md / 6; int sd = md % 6;
  int n = nt * 16 + (lane & 15);
  int k = kt * 32 + ((lane >> 4) << 3) + (e & 7) + ((e >> 3) << 4);
  const float* W = (mat ? whh : wih) + ((size_t)sd * 768 + n) * 256 + k;
  ((__bf16*)dst)[p] = (__bf16)(*W);
}

// 4b) max-abs reduction (fp bits compare valid for non-negative floats)
__global__ void k_maxabs(const float* src, int n, unsigned* dst) {
  int i = blockIdx.x * blockDim.x + threadIdx.x;
  float v = 0.f;
  for (; i < n; i += gridDim.x * blockDim.x) { float a = fabsf(src[i]); v = a > v ? a : v; }
  atomicMax(dst, __float_as_uint(v));
}

// 4c) fake-quantize + pack into B-tile layout with zero padding
__global__ void k_packQ(const float* src, int Nrows, int Kcols, int Nt, int Kt,
                        const unsigned* mxbits, unsigned short* dst) {
  size_t total = (size_t)Nt * Kt * 512;
  size_t p = (size_t)blockIdx.x * blockDim.x + threadIdx.x;
  if (p >= total) return;
  int e = p & 15; int lane = (p >> 4) & 31; size_t tix = p >> 9;
  int kt = (int)(tix % Kt); int nt = (int)(tix / Kt);
  int n = nt * 16 + (lane & 15);
  int k = kt * 32 + ((lane >> 4) << 3) + (e & 7) + ((e >> 3) << 4);
  float v = 0.f;
  if (n < Nrows && k < Kcols) {
    float s = __uint_as_float(*mxbits) / 127.f; s = s > 1e-8f ? s : 1e-8f;
    float w = src[(size_t)n * Kcols + k];
    float q = rintf(w / s); q = q > 127.f ? 127.f : (q < -128.f ? -128.f : q);
    v = q * s;
  }
  ((__bf16*)dst)[p] = (__bf16)v;
}

// ---------------------------------------------------------------------------
// 5) bidirectional GRU: one block per (scale, dir, m-tile). 8 waves own 2
//    hidden column-tiles each; h round-trips through LDS each step.
// ---------------------------------------------------------------------------
__launch_bounds__(256)
__global__ void k_gru(const unsigned short* x, const unsigned short* gruW,
                      const float* bih, const float* bhh, const int* cnts,
                      unsigned short* feat) {
  __shared__ __bf16 hlds[16 * 256];
  __shared__ int    effs[16];

  int bid = blockIdx.x;
  int si, dir, mt;
  if (bid < 2)       { si = 0; dir = bid;             mt = 0; }
  else if (bid < 8)  { si = 1; int r = bid - 2; dir = r / 3; mt = r % 3; }
  else               { si = 2; dir = bid - 8;          mt = 0; }
  int ioff    = si == 0 ? 0  : (si == 1 ? 15 : 55);
  int I       = si == 1 ? 40 : 15;
  int rowBase = si == 0 ? 0  : (si == 1 ? 16 : 64);
  int sd      = si * 2 + dir;

  int tid  = threadIdx.x;
  int lane = tid & 31;
  int wv   = tid >> 5;        // wave 0..7
  int half = lane >> 4;
  int lcol = lane & 15;
  int kb   = half ? 8 : 0;    // A-fragment K base within 32-chunk
  int myrow = lcol;           // A row owned by this lane

  for (int i = tid; i < 16 * 256; i += 256) hlds[i] = (__bf16)0.f;
  if (tid < 16) {
    int g = ioff + mt * 16 + tid;
    if (g >= ioff + I) g = ioff + I - 1;
    int c = cnts[g];
    effs[tid] = c < 1 ? 1 : c;
  }
  __syncthreads();

  int g_me = ioff + mt * 16 + myrow; if (g_me >= ioff + I) g_me = ioff + I - 1;
  int eff_me = effs[myrow];
  int effA[8];
  #pragma unroll
  for (int j = 0; j < 8; ++j) effA[j] = effs[j + 8 * half];

  float bI[2][3], bH[2][3];
  #pragma unroll
  for (int c = 0; c < 2; ++c) {
    int ct = wv * 2 + c;
    #pragma unroll
    for (int g = 0; g < 3; ++g) {
      int col = g * 256 + ct * 16 + lcol;
      bI[c][g] = bih[(size_t)sd * 768 + col];
      bH[c][g] = bhh[(size_t)sd * 768 + col];
    }
  }

  const unsigned short* wihB = gruW + (size_t)sd * 196608;
  const unsigned short* whhB = gruW + (size_t)(6 + sd) * 196608;

  v8f hc[2]; hc[0] = zero8(); hc[1] = zero8();
  const size_t xrowbase = (size_t)g_me * MAXT * DIM;

  #pragma unroll 1
  for (int t = 0; t < MAXT; ++t) {
    int tx = dir ? (eff_me - 1 - t) : t;
    tx = tx < 0 ? 0 : (tx > MAXT - 1 ? MAXT - 1 : tx);
    const unsigned short* xrow = x + xrowbase + (size_t)tx * DIM;

    // prefetch next step's embedding row (global_prefetch_b8)
    {
      int tn = dir ? (eff_me - 2 - t) : (t + 1);
      tn = tn < 0 ? 0 : (tn > MAXT - 1 ? MAXT - 1 : tn);
      __builtin_prefetch(x + xrowbase + (size_t)tn * DIM + kb, 0, 1);
    }

    v8f accI[2][3], accH[2][3];
    #pragma unroll
    for (int c = 0; c < 2; ++c)
      #pragma unroll
      for (int g = 0; g < 3; ++g) { accI[c][g] = zero8(); accH[c][g] = zero8(); }

    #pragma unroll 2
    for (int kk = 0; kk < 8; ++kk) {
      v8bf axlo = *(const v8bf*)((const __bf16*)xrow + kk * 32 + kb);
      v8bf axhi = *(const v8bf*)((const __bf16*)xrow + kk * 32 + kb + 16);
      v16bf ax = join16(axlo, axhi);
      v8bf ahlo = *(const v8bf*)(&hlds[myrow * 256 + kk * 32 + kb]);
      v8bf ahhi = *(const v8bf*)(&hlds[myrow * 256 + kk * 32 + kb + 16]);
      v16bf ah = join16(ahlo, ahhi);
      #pragma unroll
      for (int c = 0; c < 2; ++c) {
        int ct = wv * 2 + c;
        #pragma unroll
        for (int g = 0; g < 3; ++g) {
          int nt = g * 16 + ct;   // column tile within 48 (768 outputs)
          const __bf16* bp = (const __bf16*)wihB + ((size_t)(nt * 8 + kk) * 32 + lane) * 16;
          v16bf bi = join16(*(const v8bf*)bp, *(const v8bf*)(bp + 8));
          const __bf16* hp = (const __bf16*)whhB + ((size_t)(nt * 8 + kk) * 32 + lane) * 16;
          v16bf bh = join16(*(const v8bf*)hp, *(const v8bf*)(hp + 8));
          accI[c][g] = __builtin_amdgcn_wmma_f32_16x16x32_bf16(
              false, ax, false, bi, (short)0, accI[c][g], false, false);
          accH[c][g] = __builtin_amdgcn_wmma_f32_16x16x32_bf16(
              false, ah, false, bh, (short)0, accH[c][g], false, false);
        }
      }
    }

    // gate math (fp32, C layout: elem j -> row j+8*half, col = ct*16+lcol)
    #pragma unroll
    for (int c = 0; c < 2; ++c) {
      #pragma unroll
      for (int j = 0; j < 8; ++j) {
        float ir = accI[c][0][j] + bI[c][0];
        float hr = accH[c][0][j] + bH[c][0];
        float iz = accI[c][1][j] + bI[c][1];
        float hz = accH[c][1][j] + bH[c][1];
        float in_ = accI[c][2][j] + bI[c][2];
        float hn  = accH[c][2][j] + bH[c][2];
        float r = fsig(ir + hr);
        float z = fsig(iz + hz);
        float n = ftanh(in_ + r * hn);
        float hnew = (1.f - z) * n + z * hc[c][j];
        if (t < effA[j]) hc[c][j] = hnew;
      }
    }
    __syncthreads();
    #pragma unroll
    for (int c = 0; c < 2; ++c) {
      int ct = wv * 2 + c;
      #pragma unroll
      for (int j = 0; j < 8; ++j)
        hlds[(j + 8 * half) * 256 + ct * 16 + lcol] = (__bf16)hc[c][j];
    }
    __syncthreads();
  }

  // write final h rows to the feature buffer (cols dir*256 .. +255)
  #pragma unroll
  for (int c = 0; c < 2; ++c) {
    int ct = wv * 2 + c;
    #pragma unroll
    for (int j = 0; j < 8; ++j) {
      int r = j + 8 * half;
      if (mt * 16 + r < I)
        ((__bf16*)feat)[(size_t)(rowBase + mt * 16 + r) * KPAD + dir * 256 + ct * 16 + lcol]
            = (__bf16)hc[c][j];
    }
  }
}

// ---------------------------------------------------------------------------
// 6) MLP head: 519->256 relu ->256 relu -> out, all bf16 WMMA, LDS relayout
// ---------------------------------------------------------------------------
__launch_bounds__(256)
__global__ void k_mlp(const unsigned short* feat, const unsigned short* w1q,
                      const unsigned short* w2q, const unsigned short* w3q,
                      float* outbuf) {
  __shared__ __bf16 h1[16 * 256];
  __shared__ __bf16 h2[16 * 256];
  int bid = blockIdx.x;  // 0..4
  int si, mt;
  if (bid == 0)      { si = 0; mt = 0; }
  else if (bid < 4)  { si = 1; mt = bid - 1; }
  else               { si = 2; mt = 0; }
  int rowBase = si == 0 ? 0 : (si == 1 ? 16 : 64);
  int Nt3 = (si == 1) ? 8 : 11;

  int tid = threadIdx.x, lane = tid & 31, wv = tid >> 5;
  int half = lane >> 4, lcol = lane & 15;
  int kb = half ? 8 : 0;
  int myrow = lcol;

  const __bf16* W1 = (const __bf16*)w1q + (size_t)si * 16 * 17 * 512;
  const __bf16* W2 = (const __bf16*)w2q + (size_t)si * 16 * 8 * 512;
  const __bf16* W3 = (const __bf16*)w3q + (size_t)si * 11 * 8 * 512;
  const __bf16* arow = (const __bf16*)feat + (size_t)(rowBase + mt * 16 + myrow) * KPAD;

  v8f acc[2]; acc[0] = zero8(); acc[1] = zero8();
  #pragma unroll 1
  for (int kk = 0; kk < 17; ++kk) {
    v16bf a = join16(*(const v8bf*)(arow + kk * 32 + kb),
                     *(const v8bf*)(arow + kk * 32 + kb + 16));
    #pragma unroll
    for (int c = 0; c < 2; ++c) {
      int nt = wv * 2 + c;
      const __bf16* bp = W1 + ((size_t)(nt * 17 + kk) * 32 + lane) * 16;
      v16bf b = join16(*(const v8bf*)bp, *(const v8bf*)(bp + 8));
      acc[c] = __builtin_amdgcn_wmma_f32_16x16x32_bf16(false, a, false, b, (short)0, acc[c], false, false);
    }
  }
  #pragma unroll
  for (int c = 0; c < 2; ++c) {
    int ct = wv * 2 + c;
    #pragma unroll
    for (int j = 0; j < 8; ++j) {
      float v = acc[c][j]; v = v > 0.f ? v : 0.f;
      h1[(j + 8 * half) * 256 + ct * 16 + lcol] = (__bf16)v;
    }
  }
  __syncthreads();

  acc[0] = zero8(); acc[1] = zero8();
  #pragma unroll 1
  for (int kk = 0; kk < 8; ++kk) {
    v16bf a = join16(*(const v8bf*)(&h1[myrow * 256 + kk * 32 + kb]),
                     *(const v8bf*)(&h1[myrow * 256 + kk * 32 + kb + 16]));
    #pragma unroll
    for (int c = 0; c < 2; ++c) {
      int nt = wv * 2 + c;
      const __bf16* bp = W2 + ((size_t)(nt * 8 + kk) * 32 + lane) * 16;
      v16bf b = join16(*(const v8bf*)bp, *(const v8bf*)(bp + 8));
      acc[c] = __builtin_amdgcn_wmma_f32_16x16x32_bf16(false, a, false, b, (short)0, acc[c], false, false);
    }
  }
  #pragma unroll
  for (int c = 0; c < 2; ++c) {
    int ct = wv * 2 + c;
    #pragma unroll
    for (int j = 0; j < 8; ++j) {
      float v = acc[c][j]; v = v > 0.f ? v : 0.f;
      h2[(j + 8 * half) * 256 + ct * 16 + lcol] = (__bf16)v;
    }
  }
  __syncthreads();

  #pragma unroll 1
  for (int pass = 0; pass < 2; ++pass) {
    int nt = wv + pass * 8;
    if (nt >= Nt3) continue;
    v8f a3 = zero8();
    #pragma unroll 1
    for (int kk = 0; kk < 8; ++kk) {
      v16bf a = join16(*(const v8bf*)(&h2[myrow * 256 + kk * 32 + kb]),
                       *(const v8bf*)(&h2[myrow * 256 + kk * 32 + kb + 16]));
      const __bf16* bp = W3 + ((size_t)(nt * 8 + kk) * 32 + lane) * 16;
      v16bf b = join16(*(const v8bf*)bp, *(const v8bf*)(bp + 8));
      a3 = __builtin_amdgcn_wmma_f32_16x16x32_bf16(false, a, false, b, (short)0, a3, false, false);
    }
    #pragma unroll
    for (int j = 0; j < 8; ++j)
      outbuf[(size_t)(rowBase + mt * 16 + j + 8 * half) * NPAD3 + nt * 16 + lcol] = a3[j];
  }
}

// ---------------------------------------------------------------------------
// 7) finalize: softmax-weighted offsets, clip, assemble output
// ---------------------------------------------------------------------------
__global__ void k_final(const float* outbuf, const float* anchors, const float* alen,
                        const float* sw0, const float* ew0, const float* sw1,
                        const float* ew1, const float* sw2, const float* ew2,
                        float* out) {
  __shared__ float red[128];
  int i = blockIdx.x; int tid = threadIdx.x;
  int si = i < 15 ? 0 : (i < 55 ? 1 : 2);
  int li = i - (si == 0 ? 0 : (si == 1 ? 15 : 55));
  int rowBase = si == 0 ? 0 : (si == 1 ? 16 : 64);
  int b = si == 1 ? 60 : 80;
  const float* row = outbuf + (size_t)(rowBase + li) * NPAD3;
  const float* sw = si == 0 ? sw0 : (si == 1 ? sw1 : sw2);
  const float* ew = si == 0 ? ew0 : (si == 1 ? ew1 : ew2);
  float al = alen[0];
  float s = anchors[2*i], e = anchors[2*i+1];

  float so = 0.f, eo = 0.f;
  for (int which = 0; which < 2; ++which) {
    const float* src  = row + which * b;
    const float* wvec = which ? ew : sw;
    float v = (tid < b) ? src[tid] : -1e30f;
    red[tid] = v; __syncthreads();
    for (int st = 64; st > 0; st >>= 1) {
      if (tid < st) { float o2 = red[tid + st]; if (o2 > red[tid]) red[tid] = o2; }
      __syncthreads();
    }
    float mx = red[0]; __syncthreads();
    float ex = (tid < b) ? __expf(v - mx) : 0.f;
    red[tid] = ex; __syncthreads();
    for (int st = 64; st > 0; st >>= 1) { if (tid < st) red[tid] += red[tid + st]; __syncthreads(); }
    float sum = red[0]; __syncthreads();
    red[tid] = (tid < b) ? ex * wvec[tid] : 0.f; __syncthreads();
    for (int st = 64; st > 0; st >>= 1) { if (tid < st) red[tid] += red[tid + st]; __syncthreads(); }
    float dot = red[0] / sum; __syncthreads();
    if (which == 0) so = dot; else eo = dot;
  }
  if (tid == 0) {
    float b0 = s + so; b0 = b0 < 0.f ? 0.f : (b0 > al ? al : b0);
    float b1 = e + eo; b1 = b1 < 0.f ? 0.f : (b1 > al ? al : b1);
    out[2 * i] = b0; out[2 * i + 1] = b1;
    out[140 + i] = row[2 * b];
    for (int c = 0; c < 4; ++c) out[210 + 4 * i + c] = row[2 * b + 1 + c];
  }
}

// ---------------------------------------------------------------------------
// launch
// ---------------------------------------------------------------------------
extern "C" void kernel_launch(void* const* d_in, const int* in_sizes, int n_in,
                              void* d_out, int out_size, void* d_ws, size_t ws_size,
                              hipStream_t stream) {
  (void)in_sizes; (void)n_in; (void)out_size; (void)ws_size;
  const float* emb   = (const float*)d_in[0];
  const float* tp    = (const float*)d_in[1];
  const float* npred = (const float*)d_in[2];
  const float* alen  = (const float*)d_in[3];
  const float* anch  = (const float*)d_in[4];
  const float* wih   = (const float*)d_in[5];
  const float* whh   = (const float*)d_in[6];
  const float* bih   = (const float*)d_in[7];
  const float* bhh   = (const float*)d_in[8];
  const float* w1    = (const float*)d_in[9];
  const float* w2    = (const float*)d_in[10];
  const float* w3s[3] = {(const float*)d_in[11], (const float*)d_in[12], (const float*)d_in[13]};
  const float* sw0 = (const float*)d_in[14];
  const float* ew0 = (const float*)d_in[15];
  const float* sw1 = (const float*)d_in[16];
  const float* ew1 = (const float*)d_in[17];
  const float* sw2 = (const float*)d_in[18];
  const float* ew2 = (const float*)d_in[19];

  char* ws = (char*)d_ws;
  size_t off = 0;
  auto A = [&](size_t bytes) { size_t r = off; off = (off + bytes + 255) & ~(size_t)255; return r; };
  int*            idxbuf = (int*)(ws + A((size_t)70 * MAXT * 4));
  int*            cnts   = (int*)(ws + A(70 * 4));
  unsigned short* xbuf   = (unsigned short*)(ws + A((size_t)70 * MAXT * DIM * 2));
  unsigned short* gruW   = (unsigned short*)(ws + A((size_t)12 * 768 * 256 * 2));
  unsigned short* w1q    = (unsigned short*)(ws + A((size_t)3 * 16 * 17 * 512 * 2));
  unsigned short* w2q    = (unsigned short*)(ws + A((size_t)3 * 16 * 8 * 512 * 2));
  unsigned short* w3q    = (unsigned short*)(ws + A((size_t)3 * 11 * 8 * 512 * 2));
  unsigned short* featb  = (unsigned short*)(ws + A((size_t)80 * KPAD * 2));
  float*          outbuf = (float*)(ws + A((size_t)80 * NPAD3 * 4));
  unsigned*       mx     = (unsigned*)(ws + A(9 * 4));

  k_init<<<(80 * KPAD + 255) / 256, 256, 0, stream>>>(featb, mx);
  k_prep<<<70, 256, 0, stream>>>(tp, npred, alen, anch, idxbuf, cnts, featb);
  k_gather<<<70 * MAXT, 256, 0, stream>>>(emb, idxbuf, cnts, xbuf);
  k_packGru<<<(int)(((size_t)12 * 768 * 256 + 255) / 256), 256, 0, stream>>>(wih, whh, gruW);

  for (int si = 0; si < 3; ++si) {
    int n3 = (si == 1 ? 125 : 165);
    k_maxabs<<<64, 256, 0, stream>>>(w1 + (size_t)si * 256 * 519, 256 * 519, mx + si);
    k_maxabs<<<64, 256, 0, stream>>>(w2 + (size_t)si * 256 * 256, 256 * 256, mx + 3 + si);
    k_maxabs<<<64, 256, 0, stream>>>(w3s[si], n3 * 256, mx + 6 + si);
  }
  for (int si = 0; si < 3; ++si) {
    int n3 = (si == 1 ? 125 : 165);
    k_packQ<<<(16 * 17 * 512 + 255) / 256, 256, 0, stream>>>(
        w1 + (size_t)si * 256 * 519, 256, 519, 16, 17, mx + si,
        w1q + (size_t)si * 16 * 17 * 512);
    k_packQ<<<(16 * 8 * 512 + 255) / 256, 256, 0, stream>>>(
        w2 + (size_t)si * 256 * 256, 256, 256, 16, 8, mx + 3 + si,
        w2q + (size_t)si * 16 * 8 * 512);
    k_packQ<<<(11 * 8 * 512 + 255) / 256, 256, 0, stream>>>(
        w3s[si], n3, 256, 11, 8, mx + 6 + si,
        w3q + (size_t)si * 11 * 8 * 512);
  }

  k_gru<<<10, 256, 0, stream>>>(xbuf, gruW, bih, bhh, cnts, featb);
  k_mlp<<<5, 256, 0, stream>>>(featb, w1q, w2q, w3q, outbuf);
  k_final<<<70, 128, 0, stream>>>(outbuf, anch, alen, sw0, ew0, sw1, ew1, sw2, ew2,
                                  (float*)d_out);
}
